// MultiHeadedAttention_42185168781559
// MI455X (gfx1250) — matrix operand order repaired and checked by reading, not compiled, not run on hardware
//
#include <hip/hip_runtime.h>
#include <hip/hip_bf16.h>

// ---------------------------------------------------------------------------
// MultiHeadedAttention for MI455X (gfx1250), bf16 WMMA pipeline.
// B=2, S=2048, D=1024, H=16, K=64.  Softmax over the QUERY axis (columns of
// the SxS score matrix):  ctx[s,:] = sum_t exp(q_s.k_t/8) * (v[t,:]/Z_t),
// Z_t = sum_s exp(q_s.k_t/8).  Two-pass: stats pass for 1/Z, fused recompute
// pass for ctx.  Scores never touch HBM.  bf16 tile staging uses
// GLOBAL_LOAD_ASYNC_TO_LDS_B128 (ASYNCcnt) instead of load+ds_store.
// ---------------------------------------------------------------------------

#define BB 2
#define SS 2048
#define DD 1024
#define HH 16
#define KD 64

typedef __attribute__((ext_vector_type(16))) __bf16 v16bf;
typedef __attribute__((ext_vector_type(8)))  float  v8f;

// --- CDNA5 async global->LDS copy (16B per lane), tracked by ASYNCcnt ------
__device__ inline void async_ld_lds_b128(void* lds_ptr, const void* gptr) {
  // VDST = LDS byte offset (low 32 bits of generic shared pointer),
  // VADDR = 64-bit global address, SADDR = off.
  unsigned       loff = (unsigned)(unsigned long long)(uintptr_t)lds_ptr;
  unsigned long long g = (unsigned long long)(uintptr_t)gptr;
  asm volatile("global_load_async_to_lds_b128 %0, %1, off"
               :: "v"(loff), "v"(g)
               : "memory");
}
__device__ inline void wait_async0() {
#if __has_builtin(__builtin_amdgcn_s_wait_asynccnt)
  __builtin_amdgcn_s_wait_asynccnt(0);
#else
  asm volatile("s_wait_asynccnt 0x0" ::: "memory");
#endif
}

// Stage a 128x64 bf16 tile (row stride 64 elems) global -> LDS, async.
__device__ inline void stage_tile_128x64_async(const __bf16* __restrict__ g,
                                               __bf16* lds, int tid) {
  for (int i = tid; i < 128 * 8; i += 256) {  // 1024 x 16B transfers
    int r = i >> 3, cg = i & 7;
    async_ld_lds_b128(lds + r * 64 + cg * 8, g + (size_t)r * KD + cg * 8);
  }
}

// --- WMMA fragment loaders (wave32 layouts from CDNA5 ISA 7.12.2) ----------
__device__ inline v16bf load_frag_a(const __bf16* lds, int ldk, int row_base) {
  const int lane = threadIdx.x & 31;
  const int r  = row_base + (lane & 15);
  const int kb = (lane < 16) ? 0 : 8;
  union { uint4 u[2]; v16bf v; } f;
  f.u[0] = *(const uint4*)(lds + (size_t)r * ldk + kb);
  f.u[1] = *(const uint4*)(lds + (size_t)r * ldk + kb + 16);
  return f.v;
}
__device__ inline v16bf load_frag_b(const __bf16* lds, int ldk, int col_base) {
  const int lane = threadIdx.x & 31;
  const int c  = col_base + (lane & 15);
  const int kb = (lane < 16) ? 0 : 8;
  union { uint4 u[2]; v16bf v; } f;
  f.u[0] = *(const uint4*)(lds + (size_t)c * ldk + kb);
  f.u[1] = *(const uint4*)(lds + (size_t)c * ldk + kb + 16);
  return f.v;
}

__device__ inline v8f wmma_bf16(v16bf a, v16bf b, v8f c) {
  return __builtin_amdgcn_wmma_f32_16x16x32_bf16(
      false, a, false, b, (short)0, c, false, false);
}

// ---------------------------------------------------------------------------
// Kernel 1: per-head projection.  Y[b,h,s,k] = sum_d X[b,s,d]*W[h*64+k,d]+bias
// fp32 sources are converted to bf16 while staging (VALU path).
// ---------------------------------------------------------------------------
__global__ __launch_bounds__(256)
void mha_proj_qkv(const float* __restrict__ X, const float* __restrict__ W,
                  const float* __restrict__ bias, __bf16* __restrict__ Y) {
  __shared__ __bf16 Al[128 * 32];
  __shared__ __bf16 Bl[128 * 32];
  const int tid  = threadIdx.x;
  const int lane = tid & 31;
  const int w    = tid >> 5;
  const int m0   = blockIdx.y * 128;
  const int n0   = blockIdx.x * 128;
  const int rb   = (w & 3) * 32;
  const int cb   = (w >> 2) * 64;
  v8f acc[2][4] = {};

  const int cg = tid & 7;
  const int r0 = tid >> 3;

  for (int k0 = 0; k0 < DD; k0 += 32) {
    for (int rr = r0; rr < 128; rr += 32) {
      float4 f = *(const float4*)(X + (size_t)(m0 + rr) * DD + k0 + cg * 4);
      union { __bf16 h[4]; uint2 u; } p;
      p.h[0] = (__bf16)f.x; p.h[1] = (__bf16)f.y;
      p.h[2] = (__bf16)f.z; p.h[3] = (__bf16)f.w;
      *(uint2*)&Al[rr * 32 + cg * 4] = p.u;
    }
    for (int nn = r0; nn < 128; nn += 32) {
      float4 f = *(const float4*)(W + (size_t)(n0 + nn) * DD + k0 + cg * 4);
      union { __bf16 h[4]; uint2 u; } p;
      p.h[0] = (__bf16)f.x; p.h[1] = (__bf16)f.y;
      p.h[2] = (__bf16)f.z; p.h[3] = (__bf16)f.w;
      *(uint2*)&Bl[nn * 32 + cg * 4] = p.u;
    }
    if (k0 + 32 < DD) {
      __builtin_prefetch(X + (size_t)(m0 + r0) * DD + k0 + 32, 0, 0);
      __builtin_prefetch(W + (size_t)(n0 + r0) * DD + k0 + 32, 0, 0);
    }
    __syncthreads();
    v16bf a0 = load_frag_a(Al, 32, rb);
    v16bf a1 = load_frag_a(Al, 32, rb + 16);
#pragma unroll
    for (int ni = 0; ni < 4; ++ni) {
      v16bf b = load_frag_b(Bl, 32, cb + ni * 16);
      acc[0][ni] = wmma_bf16(a0, b, acc[0][ni]);
      acc[1][ni] = wmma_bf16(a1, b, acc[1][ni]);
    }
    __syncthreads();
  }
#pragma unroll
  for (int mi = 0; mi < 2; ++mi)
#pragma unroll
    for (int ni = 0; ni < 4; ++ni)
#pragma unroll
      for (int i = 0; i < 8; ++i) {
        int row = m0 + rb + mi * 16 + i + ((lane >= 16) ? 8 : 0);
        int col = n0 + cb + ni * 16 + (lane & 15);
        float v = acc[mi][ni][i] + bias[col];
        int b = row >> 11, s = row & (SS - 1);
        int h = col >> 6,  k = col & (KD - 1);
        Y[((((size_t)b * HH + h) * SS) + s) * KD + k] = (__bf16)v;
      }
}

// ---------------------------------------------------------------------------
// Kernel 2: column stats.  rZ[b,h,t] = 1 / sum_s exp(q_s . k_t / 8).
// Q/K tiles staged with async global->LDS DMA.
// ---------------------------------------------------------------------------
__global__ __launch_bounds__(256)
void mha_attn_stats(const __bf16* __restrict__ qh, const __bf16* __restrict__ kh,
                    float* __restrict__ rZ) {
  __shared__ __bf16 Kl[128 * 64];   // [t][k]
  __shared__ __bf16 Ql[128 * 64];   // [s][k]
  __shared__ float  red[4 * 128];
  const int tid  = threadIdx.x;
  const int lane = tid & 31;
  const int w    = tid >> 5;
  const int rb   = (w & 3) * 32;
  const int cb   = (w >> 2) * 64;
  const int bh   = blockIdx.y;
  const int t0   = blockIdx.x * 128;
  const __bf16* q = qh + (size_t)bh * SS * KD;
  const __bf16* k = kh + (size_t)bh * SS * KD;

  stage_tile_128x64_async(k + (size_t)t0 * KD, Kl, tid);  // K tile, once

  float colsum = 0.f;  // tid<128 owns column t0+tid
  for (int s0 = 0; s0 < SS; s0 += 128) {
    __syncthreads();  // Ql / red reuse
    stage_tile_128x64_async(q + (size_t)s0 * KD, Ql, tid);
    wait_async0();    // covers Kl on the first pass too
    __syncthreads();
    v8f acc[2][4] = {};
#pragma unroll
    for (int kk = 0; kk < KD; kk += 32) {
      v16bf a0 = load_frag_a(Ql + kk, 64, rb);
      v16bf a1 = load_frag_a(Ql + kk, 64, rb + 16);
#pragma unroll
      for (int ni = 0; ni < 4; ++ni) {
        v16bf b = load_frag_b(Kl + kk, 64, cb + ni * 16);
        acc[0][ni] = wmma_bf16(a0, b, acc[0][ni]);
        acc[1][ni] = wmma_bf16(a1, b, acc[1][ni]);
      }
    }
#pragma unroll
    for (int ni = 0; ni < 4; ++ni) {
      float sum = 0.f;
#pragma unroll
      for (int mi = 0; mi < 2; ++mi)
#pragma unroll
        for (int i = 0; i < 8; ++i)
          sum += __expf(acc[mi][ni][i] * 0.125f);
      sum += __shfl_xor(sum, 16, 32);  // join row-halves (lanes l <-> l+16)
      if (lane < 16) red[(w & 3) * 128 + cb + ni * 16 + lane] = sum;
    }
    __syncthreads();
    if (tid < 128)
      colsum += red[tid] + red[128 + tid] + red[256 + tid] + red[384 + tid];
  }
  if (tid < 128)
    rZ[(size_t)bh * SS + t0 + tid] = 1.0f / colsum;
}

// ---------------------------------------------------------------------------
// Kernel 3: fused ctx.  Recompute 128x128 score tile, P=exp(sc/8) -> LDS,
// ctx += P @ (V/Z).  Q/K tiles async-staged; V^T stage folds in 1/Z (VALU).
// ---------------------------------------------------------------------------
__global__ __launch_bounds__(256)
void mha_attn_ctx(const __bf16* __restrict__ qh, const __bf16* __restrict__ kh,
                  const __bf16* __restrict__ vh, const float* __restrict__ rZ,
                  __bf16* __restrict__ ctx) {
  __shared__ __bf16 Ql[128 * 64];    // [s][k]   (loaded once, async)
  __shared__ __bf16 Kl[128 * 64];    // [t][k]   (per chunk, async)
  __shared__ __bf16 Pl[128 * 128];   // [s][t]
  __shared__ __bf16 Vl[64 * 128];    // [k][t]   (B col-major, K=t)
  const int tid  = threadIdx.x;
  const int lane = tid & 31;
  const int w    = tid >> 5;
  const int rb   = (w & 3) * 32;
  const int cb1  = (w >> 2) * 64;
  const int cb2  = (w >> 2) * 32;
  const int bh   = blockIdx.y;
  const int s0   = blockIdx.x * 128;
  const __bf16* q = qh + (size_t)bh * SS * KD;
  const __bf16* k = kh + (size_t)bh * SS * KD;
  const __bf16* v = vh + (size_t)bh * SS * KD;
  const float*  z = rZ + (size_t)bh * SS;

  stage_tile_128x64_async(q + (size_t)s0 * KD, Ql, tid);  // Q tile, once

  v8f acc2[2][2] = {};
  for (int t0 = 0; t0 < SS; t0 += 128) {
    __syncthreads();  // Kl/Vl/Pl reuse
    stage_tile_128x64_async(k + (size_t)t0 * KD, Kl, tid);
    // V^T staged with 1/Z folded in: Vl[k][t] = bf16(v[t][k] * rZ[t])
    for (int i = tid; i < KD * 128; i += 256) {
      int t = i & 127, kk = i >> 7;
      float vv = (float)v[(size_t)(t0 + t) * KD + kk] * z[t0 + t];
      Vl[i] = (__bf16)vv;
    }
    wait_async0();    // covers Ql on the first pass too
    __syncthreads();

    // ---- GEMM 1: scores 128x128 ----
    v8f acc1[2][4] = {};
#pragma unroll
    for (int kk = 0; kk < KD; kk += 32) {
      v16bf a0 = load_frag_a(Ql + kk, 64, rb);
      v16bf a1 = load_frag_a(Ql + kk, 64, rb + 16);
#pragma unroll
      for (int ni = 0; ni < 4; ++ni) {
        v16bf b = load_frag_b(Kl + kk, 64, cb1 + ni * 16);
        acc1[0][ni] = wmma_bf16(a0, b, acc1[0][ni]);
        acc1[1][ni] = wmma_bf16(a1, b, acc1[1][ni]);
      }
    }
    // P = exp(score/8) -> LDS (row-major [s][t], A-fragment friendly)
#pragma unroll
    for (int mi = 0; mi < 2; ++mi)
#pragma unroll
      for (int ni = 0; ni < 4; ++ni)
#pragma unroll
        for (int i = 0; i < 8; ++i) {
          int row = rb + mi * 16 + i + ((lane >= 16) ? 8 : 0);
          int col = cb1 + ni * 16 + (lane & 15);
          Pl[row * 128 + col] = (__bf16)__expf(acc1[mi][ni][i] * 0.125f);
        }
    __syncthreads();

    // ---- GEMM 2: ctx += P(128xT128) @ Vt(T128x64) ----
#pragma unroll
    for (int ts = 0; ts < 128; ts += 32) {
      v16bf a0 = load_frag_a(Pl + ts, 128, rb);
      v16bf a1 = load_frag_a(Pl + ts, 128, rb + 16);
#pragma unroll
      for (int ni = 0; ni < 2; ++ni) {
        v16bf b = load_frag_b(Vl + ts, 128, cb2 + ni * 16);
        acc2[0][ni] = wmma_bf16(a0, b, acc2[0][ni]);
        acc2[1][ni] = wmma_bf16(a1, b, acc2[1][ni]);
      }
    }
  }
  const int b = bh >> 4, h = bh & 15;
#pragma unroll
  for (int mi = 0; mi < 2; ++mi)
#pragma unroll
    for (int ni = 0; ni < 2; ++ni)
#pragma unroll
      for (int i = 0; i < 8; ++i) {
        int s  = s0 + rb + mi * 16 + i + ((lane >= 16) ? 8 : 0);
        int kk = cb2 + ni * 16 + (lane & 15);
        ctx[(((size_t)b * SS + s) * HH + h) * KD + kk] = (__bf16)acc2[mi][ni][i];
      }
}

// ---------------------------------------------------------------------------
// Kernel 4: output projection.  out[r,e] = sum_d ctx[r,d]*Wo[e,d] + bo[e]
// ctx tile (already bf16) staged with async DMA; Wo converted on the fly.
// ---------------------------------------------------------------------------
__global__ __launch_bounds__(256)
void mha_out_proj(const __bf16* __restrict__ A, const float* __restrict__ W,
                  const float* __restrict__ bias, float* __restrict__ out) {
  __shared__ __bf16 Al[128 * 32];
  __shared__ __bf16 Bl[128 * 32];
  const int tid  = threadIdx.x;
  const int lane = tid & 31;
  const int w    = tid >> 5;
  const int m0   = blockIdx.y * 128;
  const int n0   = blockIdx.x * 128;
  const int rb   = (w & 3) * 32;
  const int cb   = (w >> 2) * 64;
  v8f acc[2][4] = {};

  for (int k0 = 0; k0 < DD; k0 += 32) {
    // A: 128x32 bf16 tile = 512 x 16B, async DMA to LDS
    for (int i = tid; i < 128 * 4; i += 256) {
      int r = i >> 2, cg = i & 3;
      async_ld_lds_b128(&Al[r * 32 + cg * 8],
                        A + (size_t)(m0 + r) * DD + k0 + cg * 8);
    }
    {
      int cg = tid & 7, r0 = tid >> 3;
      for (int nn = r0; nn < 128; nn += 32) {
        float4 f = *(const float4*)(W + (size_t)(n0 + nn) * DD + k0 + cg * 4);
        union { __bf16 h[4]; uint2 u; } p;
        p.h[0] = (__bf16)f.x; p.h[1] = (__bf16)f.y;
        p.h[2] = (__bf16)f.z; p.h[3] = (__bf16)f.w;
        *(uint2*)&Bl[nn * 32 + cg * 4] = p.u;
      }
    }
    wait_async0();
    __syncthreads();
    v16bf a0 = load_frag_a(Al, 32, rb);
    v16bf a1 = load_frag_a(Al, 32, rb + 16);
#pragma unroll
    for (int ni = 0; ni < 4; ++ni) {
      v16bf b = load_frag_b(Bl, 32, cb + ni * 16);
      acc[0][ni] = wmma_bf16(a0, b, acc[0][ni]);
      acc[1][ni] = wmma_bf16(a1, b, acc[1][ni]);
    }
    __syncthreads();
  }
#pragma unroll
  for (int mi = 0; mi < 2; ++mi)
#pragma unroll
    for (int ni = 0; ni < 4; ++ni)
#pragma unroll
      for (int i = 0; i < 8; ++i) {
        int row = m0 + rb + mi * 16 + i + ((lane >= 16) ? 8 : 0);
        int col = n0 + cb + ni * 16 + (lane & 15);
        out[(size_t)row * DD + col] = acc[mi][ni][i] + bias[col];
      }
}

// ---------------------------------------------------------------------------
// Launch.  Workspace layout (bytes):
//   [0,8M) qh | [8M,16M) kh | [16M,24M) vh  (bf16 [B,H,S,64])
//   [24M,+256K) rZ fp32 [B,H,S] | [25M,+8M) ctx bf16 [B,S,H,64]
// ---------------------------------------------------------------------------
extern "C" void kernel_launch(void* const* d_in, const int* in_sizes, int n_in,
                              void* d_out, int out_size, void* d_ws, size_t ws_size,
                              hipStream_t stream) {
  (void)in_sizes; (void)n_in; (void)out_size; (void)ws_size;
  const float* query = (const float*)d_in[0];
  const float* key   = (const float*)d_in[1];
  const float* value = (const float*)d_in[2];
  const float* Wq = (const float*)d_in[4];
  const float* bq = (const float*)d_in[5];
  const float* Wk = (const float*)d_in[6];
  const float* bk = (const float*)d_in[7];
  const float* Wv = (const float*)d_in[8];
  const float* bv = (const float*)d_in[9];
  const float* Wo = (const float*)d_in[10];
  const float* bo = (const float*)d_in[11];
  float* out = (float*)d_out;

  char* ws = (char*)d_ws;
  __bf16* qh  = (__bf16*)(ws);
  __bf16* kh  = (__bf16*)(ws + (8u << 20));
  __bf16* vh  = (__bf16*)(ws + (16u << 20));
  float*  rZ  = (float*) (ws + (24u << 20));
  __bf16* ctx = (__bf16*)(ws + (25u << 20));

  dim3 blk(256);
  dim3 gproj(DD / 128, (BB * SS) / 128, 1);   // 8 x 32
  hipLaunchKernelGGL(mha_proj_qkv, gproj, blk, 0, stream, query, Wq, bq, qh);
  hipLaunchKernelGGL(mha_proj_qkv, gproj, blk, 0, stream, key,   Wk, bk, kh);
  hipLaunchKernelGGL(mha_proj_qkv, gproj, blk, 0, stream, value, Wv, bv, vh);

  dim3 gattn(SS / 128, BB * HH, 1);           // 16 x 32
  hipLaunchKernelGGL(mha_attn_stats, gattn, blk, 0, stream, qh, kh, rZ);
  hipLaunchKernelGGL(mha_attn_ctx,   gattn, blk, 0, stream, qh, kh, vh, rZ, ctx);

  hipLaunchKernelGGL(mha_out_proj, gproj, blk, 0, stream, ctx, Wo, bo, out);
}
